// Model_16295105921612
// MI455X (gfx1250) — matrix operand order repaired and checked by reading, not compiled, not run on hardware
//
#include <hip/hip_runtime.h>
#include <hip/hip_bf16.h>
#include <stdint.h>

typedef __attribute__((ext_vector_type(16))) __bf16 v16bf;
typedef __attribute__((ext_vector_type(8)))  float  v8f;

#define Hh 43
#define Ss 2048
#define Dd 401
#define Dp 416          // D padded to 13 chunks of 32
#define NDCH 13         // 32-wide d chunks for QK
#define NOCH 26         // 16-wide output-d chunks for PV
#define KB 32           // keys per block
#define QB 16           // queries per workgroup
#define NTHREADS 256
#define NWAVES 8
#define NKBLK (Ss / KB)
#define SCALE 0.04993761694f   // 1/sqrt(401)
#define INV_KEEP 10.0f
#define KEEP_THRESH 1677721u   // 0.1 * 2^24

__device__ __forceinline__ uint32_t hashrng(uint32_t x) {
    x *= 0x9E3779B1u; x ^= x >> 16;
    x *= 0x85EBCA6Bu; x ^= x >> 13;
    x *= 0xC2B2AE35u; x ^= x >> 16;
    return x;
}

// A-matrix fragment (16x32 bf16): lane m=lane%16 row; reg j holds K pair per ISA table.
__device__ __forceinline__ v16bf load_a16(const __bf16* rowChunk, int hlf) {
    union { v16bf v; uint32_t u[8]; } f;
    const uint32_t* p = reinterpret_cast<const uint32_t*>(rowChunk);
#pragma unroll
    for (int j = 0; j < 8; ++j) {
        int kb = ((j < 4) ? (2 * j) : (16 + 2 * (j - 4))) + 8 * hlf;
        f.u[j] = p[kb >> 1];
    }
    return f.v;
}

// B-matrix fragment (32x16 bf16): lane n=lane%16 column; reg j holds K pair per ISA table (same as A).
// colBase points at the 32 contiguous bf16 K-dim values for this lane's column.
__device__ __forceinline__ v16bf load_b16(const __bf16* colBase, int hlf) {
    union { v16bf v; uint32_t u[8]; } f;
    const uint32_t* p = reinterpret_cast<const uint32_t*>(colBase);
#pragma unroll
    for (int j = 0; j < 8; ++j) {
        int kb = ((j < 4) ? (2 * j) : (16 + 2 * (j - 4))) + 8 * hlf;
        f.u[j] = p[kb >> 1];
    }
    return f.v;
}

__global__ __launch_bounds__(NTHREADS) void fa_fwd_kernel(
    const float* __restrict__ Qg, const float* __restrict__ Kg,
    const float* __restrict__ Vg, const float* __restrict__ Mg,
    float* __restrict__ Og) {
    __shared__ __bf16 sQ[QB][Dp];        // 13312 B
    __shared__ __bf16 sK[KB][Dp];        // 26624 B
    __shared__ __bf16 sVt[Dp][KB];       // 26624 B (transposed V for B-frag pair loads)
    __shared__ __bf16 sP[NWAVES][QB][KB]; // 8192 B (per-wave P tile)

    const int h     = blockIdx.y;
    const int qbase = blockIdx.x * QB;
    const int tid   = threadIdx.x;
    const int wave  = tid >> 5;
    const int lane  = tid & 31;
    const int hlf   = lane >> 4;
    const int nn    = lane & 15;
    const size_t headOff = (size_t)h * Ss * Dd;

    // ---- stage Q tile (fp32 -> bf16, zero-pad d=401..415) ----
    for (int idx = tid; idx < QB * Dp; idx += NTHREADS) {
        int r = idx / Dp, d = idx - r * Dp;
        float v = (d < Dd) ? Qg[headOff + (size_t)(qbase + r) * Dd + d] : 0.0f;
        sQ[r][d] = (__bf16)v;
    }

    v8f acc[4] = {};
    float mreg[8], lreg[8];
#pragma unroll
    for (int r = 0; r < 8; ++r) { mreg[r] = -INFINITY; lreg[r] = 0.0f; }

    for (int kblk = 0; kblk < NKBLK; ++kblk) {
        const int kbase = kblk * KB;
        __syncthreads();  // previous block's compute done (also guards sQ on iter 0)

        // ---- stage K (row-major) and V (transposed) tiles, fp32 -> bf16 ----
        for (int idx = tid; idx < KB * Dp; idx += NTHREADS) {
            int kk = idx / Dp, d = idx - kk * Dp;
            float kv = 0.0f, vv = 0.0f;
            if (d < Dd) {
                size_t g = headOff + (size_t)(kbase + kk) * Dd + d;
                kv = Kg[g]; vv = Vg[g];
            }
            sK[kk][d]  = (__bf16)kv;
            sVt[d][kk] = (__bf16)vv;
        }
        if (kblk + 1 < NKBLK) {  // hint next block into L2 (global_prefetch_b8)
            int row = tid >> 3, seg = tid & 7;
            size_t g = headOff + (size_t)(kbase + KB + row) * Dd + seg * 56;
            __builtin_prefetch(&Kg[g], 0, 1);
            __builtin_prefetch(&Vg[g], 0, 1);
        }
        __syncthreads();

        // ---- S(16x32) = Q * K^T via bf16 WMMA over 13 d-chunks ----
        v8f S0 = {}, S1 = {};
#pragma unroll
        for (int c = 0; c < NDCH; ++c) {
            v16bf a  = load_a16(&sQ[nn][32 * c], hlf);
            v16bf b0 = load_b16(&sK[nn][32 * c], hlf);        // keys 0..15
            v16bf b1 = load_b16(&sK[nn + 16][32 * c], hlf);   // keys 16..31
            S0 = __builtin_amdgcn_wmma_f32_16x16x32_bf16(false, a, false, b0,
                                                         (short)0, S0, false, false);
            S1 = __builtin_amdgcn_wmma_f32_16x16x32_bf16(false, a, false, b1,
                                                         (short)0, S1, false, false);
        }

        // ---- scale + mask + online softmax + dropout ----
        float p0[8], p1[8], corr[8];
#pragma unroll
        for (int r = 0; r < 8; ++r) {
            const int q = qbase + r + 8 * hlf;
            const size_t mrow = (size_t)q * Ss + kbase;
            p0[r] = S0[r] * SCALE + Mg[mrow + nn];
            p1[r] = S1[r] * SCALE + Mg[mrow + 16 + nn];
        }
#pragma unroll
        for (int r = 0; r < 8; ++r) {
            float rmax = fmaxf(p0[r], p1[r]);
#pragma unroll
            for (int off = 8; off > 0; off >>= 1)
                rmax = fmaxf(rmax, __shfl_xor(rmax, off, 32));
            float mn = fmaxf(mreg[r], rmax);
            corr[r] = __expf(mreg[r] - mn);
            mreg[r] = mn;
            float e0 = __expf(p0[r] - mn);
            float e1 = __expf(p1[r] - mn);
            float rs = e0 + e1;
#pragma unroll
            for (int off = 8; off > 0; off >>= 1)
                rs += __shfl_xor(rs, off, 32);
            lreg[r] = lreg[r] * corr[r] + rs;   // denominator: pre-dropout
            // deterministic hash dropout (p=0.9 drop, 1/0.1 scale on kept)
            const int q = qbase + r + 8 * hlf;
            uint32_t base = ((uint32_t)h * Ss + (uint32_t)q) * (uint32_t)Ss + (uint32_t)kbase;
            uint32_t x0 = hashrng(base + nn);
            uint32_t x1 = hashrng(base + 16u + nn);
            p0[r] = ((x0 & 0xFFFFFFu) < KEEP_THRESH) ? e0 * INV_KEEP : 0.0f;
            p1[r] = ((x1 & 0xFFFFFFu) < KEEP_THRESH) ? e1 * INV_KEEP : 0.0f;
        }

        // ---- stash P (bf16) in this wave's private LDS tile ----
#pragma unroll
        for (int r = 0; r < 8; ++r) {
            sP[wave][r + 8 * hlf][nn]      = (__bf16)p0[r];
            sP[wave][r + 8 * hlf][nn + 16] = (__bf16)p1[r];
        }

        // ---- rescale running O by softmax correction ----
#pragma unroll
        for (int t = 0; t < 4; ++t)
#pragma unroll
            for (int r = 0; r < 8; ++r)
                acc[t][r] *= corr[r];

        // ---- O += P * V for this wave's d-chunks (wave-uniform branches: EXEC all-1) ----
        v16bf pa = load_a16(&sP[wave][nn][0], hlf);
#pragma unroll
        for (int t = 0; t < 4; ++t) {
            int ch = wave + 8 * t;
            if (ch < NOCH) {
                v16bf vb = load_b16(&sVt[16 * ch + nn][0], hlf);
                acc[t] = __builtin_amdgcn_wmma_f32_16x16x32_bf16(false, pa, false, vb,
                                                                 (short)0, acc[t], false, false);
            }
        }
    }

    // ---- normalize and store (skip pad columns) ----
#pragma unroll
    for (int t = 0; t < 4; ++t) {
        int ch = wave + 8 * t;
        if (ch < NOCH) {
            int d = 16 * ch + nn;
            if (d < Dd) {
                float invl;
#pragma unroll
                for (int r = 0; r < 8; ++r) {
                    invl = 1.0f / lreg[r];
                    int q = qbase + r + 8 * hlf;
                    Og[headOff + (size_t)q * Dd + d] = acc[t][r] * invl;
                }
            }
        }
    }
}

extern "C" void kernel_launch(void* const* d_in, const int* in_sizes, int n_in,
                              void* d_out, int out_size, void* d_ws, size_t ws_size,
                              hipStream_t stream) {
    (void)in_sizes; (void)n_in; (void)d_ws; (void)ws_size; (void)out_size;
    const float* Q = (const float*)d_in[0];
    const float* K = (const float*)d_in[1];
    const float* V = (const float*)d_in[2];
    const float* M = (const float*)d_in[3];
    float* O = (float*)d_out;
    dim3 grid(Ss / QB, Hh, 1);
    fa_fwd_kernel<<<grid, NTHREADS, 0, stream>>>(Q, K, V, M, O);
}
